// LinearCausalAttention_64793876627781
// MI455X (gfx1250) — matrix operand order repaired and checked by reading, compiled, not run on hardware
//
#include <hip/hip_runtime.h>
#include <hip/hip_bf16.h>
#include <stdint.h>

typedef __bf16 bf16_t;
typedef uint32_t u32;
typedef __attribute__((ext_vector_type(16))) __bf16 v16bf;
typedef __attribute__((ext_vector_type(8)))  __bf16 v8bf;
typedef __attribute__((ext_vector_type(8)))  float  v8f;
typedef __attribute__((ext_vector_type(4)))  u32    u32x4;
typedef __attribute__((ext_vector_type(8)))  u32    u32x8;

#define EPS_F 1e-6f
#define AS_STRIDE 40   // 32 payload halves + 8 pad halves (TDM pad: 4 DW per 16 DW)

__device__ __forceinline__ float elup(float x) { return x > 0.f ? x + 1.f : __expf(x); }

// ---------------------------------------------------------------------------
// WMMA fragment loaders (wave32, 16x16x32 bf16)
// ---------------------------------------------------------------------------
__device__ __forceinline__ v16bf load_a_frag(const bf16_t* tile, int row0, int stride, int kk) {
  int lane = threadIdx.x & 31;
  int m = lane & 15, h = lane >> 4;
  const bf16_t* p = tile + (size_t)(row0 + m) * stride + kk + 8 * h;
  v8bf lo = *(const v8bf*)p;
  v8bf hi = *(const v8bf*)(p + 16);
  v16bf r;
#pragma unroll
  for (int i = 0; i < 8; i++) { r[i] = lo[i]; r[i + 8] = hi[i]; }
  return r;
}

__device__ __forceinline__ v16bf load_b_frag(const bf16_t* tile, int row0, int stride, int kk) {
  int lane = threadIdx.x & 31;
  int n = lane & 15, h = lane >> 4;
  const bf16_t* p = tile + (size_t)(row0 + n) * stride + kk + 8 * h;
  v8bf lo = *(const v8bf*)p;
  v8bf hi = *(const v8bf*)(p + 16);
  v16bf r;
#pragma unroll
  for (int i = 0; i < 8; i++) { r[i] = lo[i]; r[i + 8] = hi[i]; }
  return r;
}

__device__ __forceinline__ v8f wmma_bf16(v16bf a, v16bf b, v8f c) {
  return __builtin_amdgcn_wmma_f32_16x16x32_bf16(false, a, false, b, (short)0, c, false, false);
}

// ---------------------------------------------------------------------------
// Tensor Data Mover: DMA a [tile_h x 32] bf16 tile (row stride row_len elems)
// into LDS with 8-half row padding (-> LDS row stride 40).  2-group D#.
// ---------------------------------------------------------------------------
__device__ __forceinline__ void tdm_load_tile(const bf16_t* gsrc, u32 lds_byte_off,
                                              u32 row_len, u32 nrows, u32 tile_h) {
  uint64_t ga = (uint64_t)(uintptr_t)gsrc;
  u32x4 g0;
  g0[0] = 1u;                                            // count=1, user mode
  g0[1] = lds_byte_off;                                  // lds_addr
  g0[2] = (u32)ga;                                       // global_addr[31:0]
  g0[3] = ((u32)(ga >> 32) & 0x01FFFFFFu) | (2u << 30);  // addr[56:32] | type=2
  u32x8 g1;
  g1[0] = (1u << 16)     // data_size = 2B
        | (1u << 20)     // pad_enable
        | (3u << 22)     // pad_interval: 16 DWORDs (one 32-half row)
        | (3u << 25);    // pad_amount: 4 DWORDs (8 halves)
  g1[1] = (row_len & 0xFFFFu) << 16;                     // tensor_dim0[15:0]
  g1[2] = (row_len >> 16) | ((nrows & 0xFFFFu) << 16);   // dim0[31:16] | dim1[15:0]
  g1[3] = (nrows >> 16) | (32u << 16);                   // dim1[31:16] | tile_dim0=32
  g1[4] = tile_h & 0xFFFFu;                              // tile_dim1
  g1[5] = row_len;                                       // tensor_dim0_stride[31:0]
  g1[6] = 0u;                                            // stride[47:32] | dim1_stride lo
  g1[7] = 0u;
  asm volatile("tensor_load_to_lds %0, %1" :: "s"(g0), "s"(g1) : "memory");
}

__device__ __forceinline__ u32 lds_off(const void* p) { return (u32)(uintptr_t)p; }

// ---------------------------------------------------------------------------
// Pre-pass: fp32 -> bf16 cast (x), and fp32 -> bf16 transpose (weights)
// ---------------------------------------------------------------------------
__global__ __launch_bounds__(256) void cast_f32_bf16_kernel(
    const float* __restrict__ in, bf16_t* __restrict__ outp, int n) {
  int i = (blockIdx.x * 256 + threadIdx.x) * 8;
  if (i + 8 <= n) {
    float fv[8];
    *(float4*)&fv[0] = *(const float4*)(in + i);
    *(float4*)&fv[4] = *(const float4*)(in + i + 4);
    bf16_t tmp[8];
#pragma unroll
    for (int j = 0; j < 8; j++) tmp[j] = (bf16_t)fv[j];
    *(uint4*)&outp[i] = *(uint4*)&tmp[0];
  }
}

__global__ __launch_bounds__(256) void transpose_f32_bf16_kernel(
    const float* __restrict__ W, bf16_t* __restrict__ Wt, int Kdim, int Ndim) {
  __shared__ float tile[32][33];
  int n0 = blockIdx.x * 32, k0 = blockIdx.y * 32;
  int tx = threadIdx.x & 31, ty = threadIdx.x >> 5;
#pragma unroll
  for (int i = 0; i < 4; i++)
    tile[ty + i * 8][tx] = W[(size_t)(k0 + ty + i * 8) * Ndim + n0 + tx];
  __syncthreads();
#pragma unroll
  for (int i = 0; i < 4; i++)
    Wt[(size_t)(n0 + ty + i * 8) * Kdim + k0 + tx] = (bf16_t)tile[tx][ty + i * 8];
}

// ---------------------------------------------------------------------------
// GEMM 1: qkv = xb @ w1t^T + b (A bf16 [M][K], B bf16 N-major [N][K]).
// TDM double-buffered main loop, one barrier per 32-K step.
// Fused bias + elu+1 on q/k, scatter to per-head [h][l][d] bf16.
// ---------------------------------------------------------------------------
__global__ __launch_bounds__(256) void qkv_gemm_kernel(
    const bf16_t* __restrict__ Ab, const bf16_t* __restrict__ Bt, const float* __restrict__ bias,
    bf16_t* __restrict__ qp, bf16_t* __restrict__ kp, bf16_t* __restrict__ vb,
    int M, int N, int K, int E, int Lseq) {
  __shared__ bf16_t As[2][128 * AS_STRIDE];
  __shared__ bf16_t Bs[2][128 * AS_STRIDE];
  const int t = threadIdx.x;
  const int wave = t >> 5;
  const int m0 = blockIdx.y * 128;
  const int n0 = blockIdx.x * 128;
  const int wm = wave >> 1;
  const int wn = wave & 1;
  const bool issuer = (wave == 0);
  v8f acc[2][4] = {};

  if (issuer) {
    tdm_load_tile(Ab + (size_t)m0 * K, lds_off(&As[0][0]), K, M, 128);
    tdm_load_tile(Bt + (size_t)n0 * K, lds_off(&Bs[0][0]), K, N, 128);
  }
  const int nk = K >> 5;
  for (int ik = 0; ik < nk; ik++) {
    const int cur = ik & 1;
    if (issuer) __builtin_amdgcn_s_wait_tensorcnt(0);
    __syncthreads();  // tile[cur] ready; all waves done reading tile[cur^1]
    if (issuer && (ik + 1) < nk) {
      int kk = (ik + 1) << 5;
      tdm_load_tile(Ab + (size_t)m0 * K + kk, lds_off(&As[cur ^ 1][0]), K, M, 128);
      tdm_load_tile(Bt + (size_t)n0 * K + kk, lds_off(&Bs[cur ^ 1][0]), K, N, 128);
    }
    v16bf a0 = load_a_frag(&As[cur][0], wm * 32, AS_STRIDE, 0);
    v16bf a1 = load_a_frag(&As[cur][0], wm * 32 + 16, AS_STRIDE, 0);
#pragma unroll
    for (int ni = 0; ni < 4; ni++) {
      v16bf b = load_b_frag(&Bs[cur][0], wn * 64 + ni * 16, AS_STRIDE, 0);
      acc[0][ni] = wmma_bf16(a0, b, acc[0][ni]);
      acc[1][ni] = wmma_bf16(a1, b, acc[1][ni]);
    }
  }

  const int lane = t & 31, nl = lane & 15, h = lane >> 4;
#pragma unroll
  for (int mi = 0; mi < 2; mi++) {
#pragma unroll
    for (int ni = 0; ni < 4; ni++) {
      int gcol = n0 + wn * 64 + ni * 16 + nl;
      float bv = bias[gcol];
      int seg = (gcol >= 2 * E) ? 2 : (gcol >= E) ? 1 : 0;   // tile-uniform
      int c = gcol - seg * E;
      int head = c >> 6, d = c & 63;
      bf16_t* dst = (seg == 0) ? qp : (seg == 1) ? kp : vb;
      bool act = (seg < 2);
#pragma unroll
      for (int r = 0; r < 8; r++) {
        int gm = m0 + wm * 32 + mi * 16 + 8 * h + r;
        float v = acc[mi][ni][r] + bv;
        if (act) v = elup(v);
        dst[((size_t)head * Lseq + gm) * 64 + d] = (bf16_t)v;
      }
    }
  }
}

// ---------------------------------------------------------------------------
// GEMM 2: out = attn(bf16) @ w2t^T + out_b (fp32 out). Same TDM main loop.
// ---------------------------------------------------------------------------
__global__ __launch_bounds__(256) void out_gemm_kernel(
    const bf16_t* __restrict__ Ab, const bf16_t* __restrict__ Bt, const float* __restrict__ bias,
    float* __restrict__ Cout, int M, int N, int K) {
  __shared__ bf16_t As[2][128 * AS_STRIDE];
  __shared__ bf16_t Bs[2][128 * AS_STRIDE];
  const int t = threadIdx.x;
  const int wave = t >> 5;
  const int m0 = blockIdx.y * 128;
  const int n0 = blockIdx.x * 128;
  const int wm = wave >> 1;
  const int wn = wave & 1;
  const bool issuer = (wave == 0);
  v8f acc[2][4] = {};

  if (issuer) {
    tdm_load_tile(Ab + (size_t)m0 * K, lds_off(&As[0][0]), K, M, 128);
    tdm_load_tile(Bt + (size_t)n0 * K, lds_off(&Bs[0][0]), K, N, 128);
  }
  const int nk = K >> 5;
  for (int ik = 0; ik < nk; ik++) {
    const int cur = ik & 1;
    if (issuer) __builtin_amdgcn_s_wait_tensorcnt(0);
    __syncthreads();
    if (issuer && (ik + 1) < nk) {
      int kk = (ik + 1) << 5;
      tdm_load_tile(Ab + (size_t)m0 * K + kk, lds_off(&As[cur ^ 1][0]), K, M, 128);
      tdm_load_tile(Bt + (size_t)n0 * K + kk, lds_off(&Bs[cur ^ 1][0]), K, N, 128);
    }
    v16bf a0 = load_a_frag(&As[cur][0], wm * 32, AS_STRIDE, 0);
    v16bf a1 = load_a_frag(&As[cur][0], wm * 32 + 16, AS_STRIDE, 0);
#pragma unroll
    for (int ni = 0; ni < 4; ni++) {
      v16bf b = load_b_frag(&Bs[cur][0], wn * 64 + ni * 16, AS_STRIDE, 0);
      acc[0][ni] = wmma_bf16(a0, b, acc[0][ni]);
      acc[1][ni] = wmma_bf16(a1, b, acc[1][ni]);
    }
  }

  const int lane = t & 31, nl = lane & 15, h = lane >> 4;
#pragma unroll
  for (int mi = 0; mi < 2; mi++) {
#pragma unroll
    for (int ni = 0; ni < 4; ni++) {
      int gcol = n0 + wn * 64 + ni * 16 + nl;
      float bv = bias[gcol];
#pragma unroll
      for (int r = 0; r < 8; r++) {
        int gm = m0 + wm * 32 + mi * 16 + 8 * h + r;
        Cout[(size_t)gm * N + gcol] = acc[mi][ni][r] + bv;
      }
    }
  }
}

// ---------------------------------------------------------------------------
// Chunked linear causal attention. 1 block (4 waves) per head, chunk T=64.
// State^T kept in registers (each wave: 16x64 strip, f32 WMMA accumulators).
// ---------------------------------------------------------------------------
#define TS 72

__global__ __launch_bounds__(128) void attn_kernel(
    const bf16_t* __restrict__ qp, const bf16_t* __restrict__ kp, const bf16_t* __restrict__ vb,
    bf16_t* __restrict__ attn, int Lseq, int E) {
  __shared__ bf16_t Qc[64 * TS];
  __shared__ bf16_t Kc[64 * TS];
  __shared__ bf16_t Kt[64 * TS];
  __shared__ bf16_t Vt[64 * TS];
  __shared__ bf16_t Pb[64 * TS];
  __shared__ bf16_t Sb[64 * TS];
  __shared__ float ksum[64];
  __shared__ float norm[64];

  const int head = blockIdx.x;
  const int t = threadIdx.x;
  const int wave = t >> 5;
  const int lane = t & 31;
  const int nl = lane & 15, hl = lane >> 4;
  const size_t hbase = (size_t)head * Lseq * 64;

  if (t < 64) ksum[t] = 0.f;
  v8f state[4] = {};
  __syncthreads();

  const int nch = Lseq / 64;
  for (int ch = 0; ch < nch; ch++) {
    const int lg = ch * 64;
    {
      int r = t >> 1, d0 = (t & 1) * 32;
      const uint4* qs = (const uint4*)(qp + hbase + (size_t)(lg + r) * 64 + d0);
      uint4 q0 = qs[0], q1 = qs[1], q2 = qs[2], q3 = qs[3];
      *(uint4*)&Qc[r * TS + d0]      = q0;
      *(uint4*)&Qc[r * TS + d0 + 8]  = q1;
      *(uint4*)&Qc[r * TS + d0 + 16] = q2;
      *(uint4*)&Qc[r * TS + d0 + 24] = q3;

      const uint4* ks = (const uint4*)(kp + hbase + (size_t)(lg + r) * 64 + d0);
      uint4 k0 = ks[0], k1 = ks[1], k2 = ks[2], k3 = ks[3];
      *(uint4*)&Kc[r * TS + d0]      = k0;
      *(uint4*)&Kc[r * TS + d0 + 8]  = k1;
      *(uint4*)&Kc[r * TS + d0 + 16] = k2;
      *(uint4*)&Kc[r * TS + d0 + 24] = k3;
      bf16_t ktmp[32];
      *(uint4*)&ktmp[0]  = k0; *(uint4*)&ktmp[8]  = k1;
      *(uint4*)&ktmp[16] = k2; *(uint4*)&ktmp[24] = k3;
#pragma unroll
      for (int i = 0; i < 32; i++) Kt[(d0 + i) * TS + r] = ktmp[i];

      const uint4* vs = (const uint4*)(vb + hbase + (size_t)(lg + r) * 64 + d0);
      uint4 v0 = vs[0], v1 = vs[1], v2 = vs[2], v3 = vs[3];
      bf16_t vtmp[32];
      *(uint4*)&vtmp[0]  = v0; *(uint4*)&vtmp[8]  = v1;
      *(uint4*)&vtmp[16] = v2; *(uint4*)&vtmp[24] = v3;
#pragma unroll
      for (int i = 0; i < 32; i++) Vt[(d0 + i) * TS + r] = vtmp[i];
    }
    {  // snapshot pre-chunk state (bf16)
      int e0 = wave * 16;
#pragma unroll
      for (int di = 0; di < 4; di++)
#pragma unroll
        for (int r = 0; r < 8; r++)
          Sb[(e0 + 8 * hl + r) * TS + di * 16 + nl] = (bf16_t)state[di][r];
    }
    __syncthreads();

    {  // Sqk = Q'K'^T, causal mask, write Pb
      int l0 = wave * 16;
      v8f s[4] = {};
      v16bf aq0 = load_a_frag(Qc, l0, TS, 0);
      v16bf aq1 = load_a_frag(Qc, l0, TS, 32);
#pragma unroll
      for (int ji = 0; ji < 4; ji++) {
        v16bf b0 = load_b_frag(Kc, ji * 16, TS, 0);
        v16bf b1 = load_b_frag(Kc, ji * 16, TS, 32);
        s[ji] = wmma_bf16(aq0, b0, s[ji]);
        s[ji] = wmma_bf16(aq1, b1, s[ji]);
      }
#pragma unroll
      for (int ji = 0; ji < 4; ji++)
#pragma unroll
        for (int r = 0; r < 8; r++) {
          int l = l0 + 8 * hl + r;
          int j = ji * 16 + nl;
          float v = (j <= l) ? s[ji][r] : 0.f;
          Pb[l * TS + j] = (bf16_t)v;
        }
    }
    __syncthreads();

    if (t < 64) {  // normalizer
      int l = t;
      float rs = 0.f;
      for (int j = 0; j < 64; j++) rs += (float)Pb[l * TS + j];
      float qk = 0.f;
      for (int d = 0; d < 64; d++) qk += (float)Qc[l * TS + d] * ksum[d];
      norm[l] = rs + qk + EPS_F;
    }
    __syncthreads();

    if (t < 64) {  // ksum += column sums of K'
      int d = t;
      float s = ksum[d];
      for (int j = 0; j < 64; j++) s += (float)Kc[j * TS + d];
      ksum[d] = s;
    }

    {  // ctx = mask(Sqk)@V + Q'@State ; /= norm ; store bf16 [l][E]
      int l0 = wave * 16;
      v8f ctx[4] = {};
      v16bf aq0 = load_a_frag(Qc, l0, TS, 0);
      v16bf aq1 = load_a_frag(Qc, l0, TS, 32);
      v16bf ap0 = load_a_frag(Pb, l0, TS, 0);
      v16bf ap1 = load_a_frag(Pb, l0, TS, 32);
#pragma unroll
      for (int di = 0; di < 4; di++) {
        v16bf bs0 = load_b_frag(Sb, di * 16, TS, 0);
        v16bf bs1 = load_b_frag(Sb, di * 16, TS, 32);
        v16bf bv0 = load_b_frag(Vt, di * 16, TS, 0);
        v16bf bv1 = load_b_frag(Vt, di * 16, TS, 32);
        ctx[di] = wmma_bf16(aq0, bs0, ctx[di]);
        ctx[di] = wmma_bf16(aq1, bs1, ctx[di]);
        ctx[di] = wmma_bf16(ap0, bv0, ctx[di]);
        ctx[di] = wmma_bf16(ap1, bv1, ctx[di]);
      }
#pragma unroll
      for (int di = 0; di < 4; di++)
#pragma unroll
        for (int r = 0; r < 8; r++) {
          int l = l0 + 8 * hl + r;
          float v = ctx[di][r] / norm[l];
          attn[(size_t)(lg + l) * E + head * 64 + di * 16 + nl] = (bf16_t)v;
        }
    }

    {  // State^T += V^T @ K'
      int e0 = wave * 16;
      v16bf av0 = load_a_frag(Vt, e0, TS, 0);
      v16bf av1 = load_a_frag(Vt, e0, TS, 32);
#pragma unroll
      for (int di = 0; di < 4; di++) {
        v16bf bk0 = load_b_frag(Kt, di * 16, TS, 0);
        v16bf bk1 = load_b_frag(Kt, di * 16, TS, 32);
        state[di] = wmma_bf16(av0, bk0, state[di]);
        state[di] = wmma_bf16(av1, bk1, state[di]);
      }
    }
    __syncthreads();
  }
}

// ---------------------------------------------------------------------------
extern "C" void kernel_launch(void* const* d_in, const int* in_sizes, int n_in,
                              void* d_out, int out_size, void* d_ws, size_t ws_size,
                              hipStream_t stream) {
  (void)n_in; (void)out_size; (void)ws_size;
  const float* x     = (const float*)d_in[0];
  const float* qkv_w = (const float*)d_in[1];
  const float* qkv_b = (const float*)d_in[2];
  const float* out_w = (const float*)d_in[3];
  const float* out_b = (const float*)d_in[4];
  float* out = (float*)d_out;

  const int E = in_sizes[2] / 3;   // 1024
  const int L = in_sizes[0] / E;   // 4096
  const int N3 = 3 * E;            // 3072
  const int H = E / 64;            // 16

  // workspace (bf16): xb[L*E] | w1t[N3*E] | w2t[E*E] | qp | kp | v | attn
  bf16_t* xb   = (bf16_t*)d_ws;
  bf16_t* w1t  = xb  + (size_t)L * E;
  bf16_t* w2t  = w1t + (size_t)N3 * E;
  bf16_t* qp   = w2t + (size_t)E * E;
  bf16_t* kp   = qp  + (size_t)E * L;
  bf16_t* vbuf = kp  + (size_t)E * L;
  bf16_t* attn = vbuf + (size_t)E * L;

  // pre-pass: cast activations, convert+transpose weights to bf16 N-major
  int nx = L * E;
  cast_f32_bf16_kernel<<<dim3((nx / 8 + 255) / 256), dim3(256), 0, stream>>>(x, xb, nx);
  transpose_f32_bf16_kernel<<<dim3(N3 / 32, E / 32), dim3(256), 0, stream>>>(qkv_w, w1t, E, N3);
  transpose_f32_bf16_kernel<<<dim3(E / 32, E / 32), dim3(256), 0, stream>>>(out_w, w2t, E, E);

  dim3 blk(256);
  qkv_gemm_kernel<<<dim3(N3 / 128, L / 128), blk, 0, stream>>>(
      xb, w1t, qkv_b, qp, kp, vbuf, L, N3, E, E, L);

  attn_kernel<<<dim3(H), dim3(128), 0, stream>>>(qp, kp, vbuf, attn, L, E);

  out_gemm_kernel<<<dim3(E / 128, L / 128), blk, 0, stream>>>(
      attn, w2t, out_b, out, L, E, E);
}
